// HyperGNNModel_74629351735416
// MI455X (gfx1250) — matrix operand order repaired and checked by reading, not compiled, not run on hardware
//
#include <hip/hip_runtime.h>
#include <hip/hip_bf16.h>
#include <math.h>

// ---------------------------------------------------------------- types ----
typedef __attribute__((ext_vector_type(16))) __bf16 v16bf;
typedef __attribute__((ext_vector_type(8)))  float  v8f;
typedef __attribute__((ext_vector_type(4)))  float  f32x4;
typedef __attribute__((ext_vector_type(4)))  int    v4i;

#define N_NODES 32768
#define N_EDGES 16384
#define NNZ     65536
#define F_IN    256
#define HID     128
#define HEADS   4
#define BATCH   64
#define TSEQ    512
#define RNN_H   256
#define GATES   1024

// typed address-space casts for the gfx1250 async-DMA builtin
// (param 0: global int4*, param 1: LDS int4*; cast via integer to avoid
//  C++ addrspace conversion rules)
#define AS1V4(p) ((__attribute__((address_space(1))) v4i*)(unsigned long long)(p))
#define AS3V4(p) ((__attribute__((address_space(3))) v4i*)(unsigned long long)(p))

__device__ inline v8f wmma_bf16(v16bf a, v16bf b, v8f c) {
  // D = A(16x32 bf16) * B(32x16 bf16) + C(16x16 f32)
  return __builtin_amdgcn_wmma_f32_16x16x32_bf16(false, a, false, b,
                                                 (short)0, c, false, false);
}

__device__ inline void wait_async0() {
#if __has_builtin(__builtin_amdgcn_s_wait_asynccnt)
  __builtin_amdgcn_s_wait_asynccnt(0);
#else
  asm volatile("s_wait_asynccnt 0x0" ::: "memory");
#endif
}

__device__ inline float sigm(float x) { return 1.f / (1.f + expf(-x)); }

__device__ inline void atomicMaxFloat(float* addr, float val) {
  if (val >= 0.f) atomicMax((int*)addr, __float_as_int(val));
  else            atomicMin((unsigned int*)addr, __float_as_uint(val));
}

// ------------------------------------------------------- generic WMMA GEMM -
// C[M,N] = act(A[M,K] * W[N,K]^T + bias[N]); M%128==0, N%128==0, K%32==0.
// Block = 256 thr = 8 waves (2x4). Each wave: 4x2 tiles of 16x16 (64x32 of C).
// Tiles staged in LDS as f32 via async DMA (b128/lane); bf16 cvt at frag build.
__global__ __launch_bounds__(256) void gemm_wmma_bf16(
    const float* __restrict__ A, const float* __restrict__ W,
    const float* __restrict__ bias, float* __restrict__ C,
    int M, int N, int K, int relu)
{
  constexpr int AST = 36;                 // padded LDS row stride (f32 elems)
  __shared__ float As[128 * AST];
  __shared__ float Ws[128 * AST];

  const int tid  = threadIdx.x;
  const int lane = tid & 31, wid = tid >> 5;
  const int wy = wid >> 2, wx = wid & 3;   // wave grid 2 (M) x 4 (N)
  const int m0 = blockIdx.y * 128, n0 = blockIdx.x * 128;
  const int g  = lane >> 4, lm = lane & 15;

  // bias: two loads per lane, issued before the K loop
  float bv[2];
#pragma unroll
  for (int b = 0; b < 2; ++b)
    bv[b] = bias ? bias[n0 + wx * 32 + b * 16 + lm] : 0.f;

  v8f acc[4][2];
  v8f vz = {};
#pragma unroll
  for (int a = 0; a < 4; ++a)
#pragma unroll
    for (int b = 0; b < 2; ++b) acc[a][b] = vz;

  for (int k0 = 0; k0 < K; k0 += 32) {
    // ---- stage 128x32 f32 A-tile and W-tile into LDS ----
#if __has_builtin(__builtin_amdgcn_global_load_async_to_lds_b128)
#pragma unroll
    for (int it = 0; it < 4; ++it) {
      int v = it * 256 + tid;             // vec4 index 0..1023
      int r = v >> 3, c4 = (v & 7) * 4;
      __builtin_amdgcn_global_load_async_to_lds_b128(
          AS1V4(&A[(size_t)(m0 + r) * K + k0 + c4]), AS3V4(&As[r * AST + c4]), 0, 0);
      __builtin_amdgcn_global_load_async_to_lds_b128(
          AS1V4(&W[(size_t)(n0 + r) * K + k0 + c4]), AS3V4(&Ws[r * AST + c4]), 0, 0);
    }
    wait_async0();
#else
    f32x4 ta[4], tw[4];
#pragma unroll
    for (int it = 0; it < 4; ++it) {
      int v = it * 256 + tid;
      int r = v >> 3, c4 = (v & 7) * 4;
      ta[it] = *(const f32x4*)&A[(size_t)(m0 + r) * K + k0 + c4];
      tw[it] = *(const f32x4*)&W[(size_t)(n0 + r) * K + k0 + c4];
    }
#pragma unroll
    for (int it = 0; it < 4; ++it) {
      int v = it * 256 + tid;
      int r = v >> 3, c4 = (v & 7) * 4;
      *(f32x4*)&As[r * AST + c4] = ta[it];
      *(f32x4*)&Ws[r * AST + c4] = tw[it];
    }
#endif
    if (k0 + 32 < K && tid < 128) {       // CDNA5 global_prefetch_b8
      __builtin_prefetch(&A[(size_t)(m0 + tid) * K + k0 + 32], 0, 0);
      __builtin_prefetch(&W[(size_t)(n0 + tid) * K + k0 + 32], 0, 0);
    }
    __syncthreads();

    // ---- fragments: vector f32 LDS reads + cvt to bf16 ----
    v16bf bfrag[2];
#pragma unroll
    for (int b = 0; b < 2; ++b) {
      const f32x4* p = (const f32x4*)&Ws[(wx * 32 + b * 16 + lm) * AST];
      f32x4 w0 = p[g * 2], w1 = p[g * 2 + 1], w2 = p[4 + g * 2], w3 = p[4 + g * 2 + 1];
#pragma unroll
      for (int j = 0; j < 4; ++j) {
        bfrag[b][j]      = (__bf16)w0[j];
        bfrag[b][4 + j]  = (__bf16)w1[j];
        bfrag[b][8 + j]  = (__bf16)w2[j];
        bfrag[b][12 + j] = (__bf16)w3[j];
      }
    }
#pragma unroll
    for (int a = 0; a < 4; ++a) {
      const f32x4* p = (const f32x4*)&As[(wy * 64 + a * 16 + lm) * AST];
      f32x4 a0 = p[g * 2], a1 = p[g * 2 + 1];        // K = g*8 .. g*8+7
      f32x4 a2 = p[4 + g * 2], a3 = p[4 + g * 2 + 1]; // K = 16+g*8 ..
      v16bf af;
#pragma unroll
      for (int j = 0; j < 4; ++j) {
        af[j]      = (__bf16)a0[j];
        af[4 + j]  = (__bf16)a1[j];
        af[8 + j]  = (__bf16)a2[j];
        af[12 + j] = (__bf16)a3[j];
      }
#pragma unroll
      for (int b = 0; b < 2; ++b) acc[a][b] = wmma_bf16(af, bfrag[b], acc[a][b]);
    }
    __syncthreads();
  }

#pragma unroll
  for (int b = 0; b < 2; ++b) {
    int n = n0 + wx * 32 + b * 16 + lm;
#pragma unroll
    for (int a = 0; a < 4; ++a)
#pragma unroll
      for (int r = 0; r < 8; ++r) {
        int m = m0 + wy * 64 + a * 16 + g * 8 + r;
        float v = acc[a][b][r] + bv[b];
        if (relu) v = fmaxf(v, 0.f);
        C[(size_t)m * N + n] = v;
      }
  }
}

// ------------------------------------------------ LSTM layer (both dirs) ---
// grid (4, 2): x = batch slice of 16 rows, y = direction (0 fwd, 1 bwd).
// 512 threads = 16 waves; wave w owns gate tiles {w, w+16, w+32, w+48} so each
// lane holds matching i/f/g/o for its (batch,hidden) element in registers.
__global__ __launch_bounds__(512) void lstm_layer(
    const float* __restrict__ preF, const float* __restrict__ preB,
    const __bf16* __restrict__ whhF, const __bf16* __restrict__ whhBk,
    float* __restrict__ out)
{
  const int dir = blockIdx.y;
  const float*  pre = dir ? preB  : preF;
  const __bf16* Whh = dir ? whhBk : whhF;
  const int colOff = dir * RNN_H;
  const int b0 = blockIdx.x * 16;

  constexpr int HS = 264;                 // padded stride for h (bf16)
  __shared__ __bf16 hbf[16 * HS];

  const int tid = threadIdx.x, lane = tid & 31, w = tid >> 5;
  const int g = lane >> 4, lm = lane & 15;
  const int jcol = w * 16 + lm;           // this lane's hidden column

  for (int i = tid; i < 16 * HS; i += 512) hbf[i] = (__bf16)0.f;
  float cst[8];
#pragma unroll
  for (int r = 0; r < 8; ++r) cst[r] = 0.f;
  v8f vz = {};
  __syncthreads();

  for (int step = 0; step < TSEQ; ++step) {
    const int t_eff = dir ? (TSEQ - 1 - step) : step;

    // hoist this step's pre[] gate loads: issued now, consumed after the
    // WMMA loop + barrier, so global latency hides behind 32 WMMAs.
    float pg[4][8];
#pragma unroll
    for (int r = 0; r < 8; ++r) {
      const size_t prow = ((size_t)(b0 + g * 8 + r) * TSEQ + t_eff) * GATES + jcol;
#pragma unroll
      for (int q = 0; q < 4; ++q) pg[q][r] = pre[prow + q * 256];
    }

    v8f acc[4];
#pragma unroll
    for (int q = 0; q < 4; ++q) acc[q] = vz;

#pragma unroll 1
    for (int kk = 0; kk < RNN_H / 32; ++kk) {
      const int kb = kk * 32;
      v16bf af;                           // A = h (16 x 256) from LDS
#pragma unroll
      for (int j = 0; j < 8; ++j) {
        af[j]     = hbf[lm * HS + kb + g * 16 + j];
        af[8 + j] = hbf[lm * HS + kb + g * 16 + 8 + j];
      }
#pragma unroll
      for (int q = 0; q < 4; ++q) {       // gate q, tile col = q*256 + w*16
        // 32B contiguous bf16 -> single vector load (2 x b128)
        v16bf bf = *(const v16bf*)&Whh[(size_t)(q * 256 + w * 16 + lm) * RNN_H
                                       + kb + g * 16];
        acc[q] = wmma_bf16(af, bf, acc[q]);
      }
    }
    __syncthreads();                      // all hbf reads done

#pragma unroll
    for (int r = 0; r < 8; ++r) {         // cell math, fully in registers
      const int m = g * 8 + r;
      float gi = acc[0][r] + pg[0][r];
      float gf = acc[1][r] + pg[1][r];
      float gg = acc[2][r] + pg[2][r];
      float go = acc[3][r] + pg[3][r];
      float c = sigm(gf) * cst[r] + sigm(gi) * tanhf(gg);
      float h = sigm(go) * tanhf(c);
      cst[r] = c;
      hbf[m * HS + jcol] = (__bf16)h;
      out[((size_t)(b0 + m) * TSEQ + t_eff) * (2 * RNN_H) + colOff + jcol] = h;
    }
    __syncthreads();
  }
}

// ------------------------------------------------------- small kernels -----
__global__ void fill_f32(float* p, float v, long long n) {
  long long i = blockIdx.x * (long long)blockDim.x + threadIdx.x;
  if (i < n) p[i] = v;
}

__global__ void convert_whh(const float* s0, const float* s1,
                            const float* s2, const float* s3, __bf16* dst) {
  int idx = blockIdx.x * blockDim.x + threadIdx.x;
  if (idx >= 4 * GATES * RNN_H) return;
  int m = idx >> 18, o = idx & (GATES * RNN_H - 1);
  const float* s = (m == 0) ? s0 : (m == 1) ? s1 : (m == 2) ? s2 : s3;
  dst[idx] = (__bf16)s[o];
}

__global__ void combine_bias(const float* a0, const float* b0,
                             const float* a1, const float* b1,
                             const float* a2, const float* b2,
                             const float* a3, const float* b3, float* dst) {
  int idx = blockIdx.x * blockDim.x + threadIdx.x;
  if (idx >= 4 * GATES) return;
  int m = idx >> 10, o = idx & (GATES - 1);
  const float* a = (m == 0) ? a0 : (m == 1) ? a1 : (m == 2) ? a2 : a3;
  const float* b = (m == 0) ? b0 : (m == 1) ? b1 : (m == 2) ? b2 : b3;
  dst[idx] = a[o] + b[o];
}

__global__ void att_dots(const float* __restrict__ xt, const float* __restrict__ ea,
                         const float* __restrict__ att_w,
                         float* __restrict__ al, float* __restrict__ ar) {
  int idx = blockIdx.x * blockDim.x + threadIdx.x;
  if (idx >= (N_NODES + N_EDGES) * HEADS) return;
  int head = idx & 3, row = idx >> 2;
  float s = 0.f;
  if (row < N_NODES) {
    const float* v = &xt[(size_t)row * (HEADS * HID) + head * HID];
    const float* a = &att_w[head * 2 * HID];
    for (int j = 0; j < HID; ++j) s += v[j] * a[j];
    al[row * HEADS + head] = s;
  } else {
    int r = row - N_NODES;
    const float* v = &ea[(size_t)r * (HEADS * HID) + head * HID];
    const float* a = &att_w[head * 2 * HID + HID];
    for (int j = 0; j < HID; ++j) s += v[j] * a[j];
    ar[r * HEADS + head] = s;
  }
}

__global__ void alpha_max(const int* __restrict__ node, const int* __restrict__ edge,
                          const float* __restrict__ al, const float* __restrict__ ar,
                          float* __restrict__ alpha, float* __restrict__ amax) {
  int idx = blockIdx.x * blockDim.x + threadIdx.x;
  if (idx >= NNZ * HEADS) return;
  int h = idx & 3, i = idx >> 2;
  float a = al[node[i] * HEADS + h] + ar[edge[i] * HEADS + h];
  a = (a > 0.f) ? a : 0.2f * a;           // leaky relu
  alpha[idx] = a;
  atomicMaxFloat(&amax[edge[i] * HEADS + h], a);
}

__global__ void alpha_expsum(const int* __restrict__ node, const int* __restrict__ edge,
                             float* __restrict__ alpha, const float* __restrict__ amax,
                             float* __restrict__ asum,
                             float* __restrict__ Dcnt, float* __restrict__ Bcnt) {
  int idx = blockIdx.x * blockDim.x + threadIdx.x;
  if (idx >= NNZ * HEADS) return;
  int h = idx & 3, i = idx >> 2;
  float e = expf(alpha[idx] - amax[edge[i] * HEADS + h]);
  alpha[idx] = e;
  atomicAdd(&asum[edge[i] * HEADS + h], e);
  if (h == 0) {
    atomicAdd(&Dcnt[node[i]], 1.f);
    atomicAdd(&Bcnt[edge[i]], 1.f);
  }
}

// block (128,4): x = hidden col, y = head; grid.x = NNZ
__global__ void scatter_edges(const int* __restrict__ node, const int* __restrict__ edge,
                              const float* __restrict__ alpha, const float* __restrict__ asum,
                              const float* __restrict__ Bcnt, const float* __restrict__ xt,
                              float* __restrict__ out_e) {
  int i = blockIdx.x, h = threadIdx.y, col = threadIdx.x;
  int e = edge[i], n = node[i];
  float bc = Bcnt[e];
  float binv = (bc > 0.f) ? 1.f / bc : 0.f;
  float an = alpha[i * HEADS + h] / (asum[e * HEADS + h] + 1e-16f);
  float val = binv * an * xt[(size_t)n * (HEADS * HID) + h * HID + col];
  atomicAdd(&out_e[((size_t)e * HEADS + h) * HID + col], val);
}

__global__ void scatter_nodes(const int* __restrict__ node, const int* __restrict__ edge,
                              const float* __restrict__ alpha, const float* __restrict__ asum,
                              const float* __restrict__ Dcnt, const float* __restrict__ out_e,
                              float* __restrict__ out_n) {
  int i = blockIdx.x, h = threadIdx.y, col = threadIdx.x;
  int e = edge[i], n = node[i];
  float dc = Dcnt[n];
  float dinv = (dc > 0.f) ? 1.f / dc : 0.f;
  float an = alpha[i * HEADS + h] / (asum[e * HEADS + h] + 1e-16f);
  float val = dinv * an * out_e[((size_t)e * HEADS + h) * HID + col];
  atomicAdd(&out_n[((size_t)n * HEADS + h) * HID + col], val);
}

__global__ void finalize_h(const float* __restrict__ out_n, const float* __restrict__ bias,
                           const float* __restrict__ ts, const float* __restrict__ first,
                           float* __restrict__ h) {
  int idx = blockIdx.x * blockDim.x + threadIdx.x;
  if (idx >= N_NODES * HID) return;
  int n = idx >> 7, k = idx & 127;
  float s = 0.f;
#pragma unroll
  for (int hd = 0; hd < HEADS; ++hd) s += out_n[((size_t)n * HEADS + hd) * HID + k];
  float v = fmaxf(s * 0.25f + bias[k], 0.f);
  float t = ts[n], f0 = first[0];
  float freq = powf(10000.f, (2.f * k) / (float)HID);
  float arg = t / freq;
  float sinus = (k & 1) ? cosf(arg) : sinf(arg);
  h[idx] = v + fmaxf(t, 0.f) + fmaxf(t - f0, 0.f) + sinus;
}

__global__ void role_add(float* __restrict__ causal, const float* __restrict__ role_emb,
                         const int* __restrict__ roles) {
  int idx = blockIdx.x * blockDim.x + threadIdx.x;
  if (idx >= N_NODES * HID) return;
  int row = idx >> 7, k = idx & 127;
  int t = row & (TSEQ - 1);
  causal[idx] += role_emb[roles[t] * HID + k];
}

// ----------------------------------------------------------------- host ----
extern "C" void kernel_launch(void* const* d_in, const int* in_sizes, int n_in,
                              void* d_out, int out_size, void* d_ws, size_t ws_size,
                              hipStream_t stream) {
  const float* x        = (const float*)d_in[0];
  const float* he_attr  = (const float*)d_in[1];
  const float* ts       = (const float*)d_in[2];
  const float* first_ts = (const float*)d_in[3];
  const float* W_lin    = (const float*)d_in[4];
  const float* att_w    = (const float*)d_in[5];
  const float* bias_cv  = (const float*)d_in[6];
  // lstm_params leaves: ((fw1,bw1),(fw2,bw2)), each (Wih,Whh,bih,bhh)
  const float* Wih1f = (const float*)d_in[7];
  const float* Whh1f = (const float*)d_in[8];
  const float* bih1f = (const float*)d_in[9];
  const float* bhh1f = (const float*)d_in[10];
  const float* Wih1b = (const float*)d_in[11];
  const float* Whh1b = (const float*)d_in[12];
  const float* bih1b = (const float*)d_in[13];
  const float* bhh1b = (const float*)d_in[14];
  const float* Wih2f = (const float*)d_in[15];
  const float* Whh2f = (const float*)d_in[16];
  const float* bih2f = (const float*)d_in[17];
  const float* bhh2f = (const float*)d_in[18];
  const float* Wih2b = (const float*)d_in[19];
  const float* Whh2b = (const float*)d_in[20];
  const float* bih2b = (const float*)d_in[21];
  const float* bhh2b = (const float*)d_in[22];
  const float* causal_W = (const float*)d_in[23];
  const float* causal_b = (const float*)d_in[24];
  const float* role_emb = (const float*)d_in[25];
  const float* emb_W    = (const float*)d_in[26];
  const float* emb_b    = (const float*)d_in[27];
  const int* he_node    = (const int*)d_in[28];
  const int* he_edge    = (const int*)d_in[29];
  const int* roles      = (const int*)d_in[30];
  float* Cout = (float*)d_out;

  // -------- workspace carve (bytes) --------
  constexpr size_t MB = 1ull << 20;
  char* W = (char*)d_ws;
  float* xt    = (float*)(W + 0);          //  64 MB (32768x512)
  float* ea    = (float*)(W + 64 * MB);    //  32 MB (16384x512)
  float* out_e = (float*)(W + 96 * MB);    //  32 MB
  float* out_n = (float*)(W + 128 * MB);   //  64 MB
  float* pre_f = (float*)(W + 0);          // 128 MB (aliases conv scratch)
  float* pre_b = (float*)(W + 128 * MB);   // 128 MB
  float* hbuf  = (float*)(W + 256 * MB);   //  16 MB (32768x128)
  float* rnn1  = (float*)(W + 272 * MB);   //  64 MB (32768x512)
  float* rnn2  = (float*)(W + 336 * MB);   //  64 MB
  float* causal= (float*)(W + 400 * MB);   //  16 MB
  float* al    = (float*)(W + 416 * MB);
  float* ar    = al + (size_t)N_NODES * HEADS;
  float* alpha = ar + (size_t)N_EDGES * HEADS;
  float* amax  = alpha + (size_t)NNZ * HEADS;
  float* asum  = amax + (size_t)N_EDGES * HEADS;
  float* Dcnt  = asum + (size_t)N_EDGES * HEADS;
  float* Bcnt  = Dcnt + N_NODES;
  __bf16* whhB = (__bf16*)(Bcnt + N_EDGES);      // 4 x 1024 x 256 bf16
  float* cbias = (float*)(whhB + 4ull * GATES * RNN_H);

  // -------- weight prep --------
  convert_whh<<<(4 * GATES * RNN_H + 255) / 256, 256, 0, stream>>>(
      Whh1f, Whh1b, Whh2f, Whh2b, whhB);
  combine_bias<<<(4 * GATES + 255) / 256, 256, 0, stream>>>(
      bih1f, bhh1f, bih1b, bhh1b, bih2f, bhh2f, bih2b, bhh2b, cbias);

  // -------- hypergraph conv --------
  gemm_wmma_bf16<<<dim3(512 / 128, N_NODES / 128), 256, 0, stream>>>(
      x, W_lin, nullptr, xt, N_NODES, 512, F_IN, 0);
  gemm_wmma_bf16<<<dim3(512 / 128, N_EDGES / 128), 256, 0, stream>>>(
      he_attr, W_lin, nullptr, ea, N_EDGES, 512, F_IN, 0);

  fill_f32<<<(N_EDGES * HEADS + 255) / 256, 256, 0, stream>>>(amax, -INFINITY, N_EDGES * HEADS);
  fill_f32<<<(N_EDGES * HEADS + 255) / 256, 256, 0, stream>>>(asum, 0.f, N_EDGES * HEADS);
  fill_f32<<<(N_NODES + 255) / 256, 256, 0, stream>>>(Dcnt, 0.f, N_NODES);
  fill_f32<<<(N_EDGES + 255) / 256, 256, 0, stream>>>(Bcnt, 0.f, N_EDGES);
  fill_f32<<<((long long)N_EDGES * HEADS * HID + 255) / 256, 256, 0, stream>>>(
      out_e, 0.f, (long long)N_EDGES * HEADS * HID);
  fill_f32<<<((long long)N_NODES * HEADS * HID + 255) / 256, 256, 0, stream>>>(
      out_n, 0.f, (long long)N_NODES * HEADS * HID);

  att_dots<<<((N_NODES + N_EDGES) * HEADS + 255) / 256, 256, 0, stream>>>(xt, ea, att_w, al, ar);
  alpha_max<<<(NNZ * HEADS + 255) / 256, 256, 0, stream>>>(he_node, he_edge, al, ar, alpha, amax);
  alpha_expsum<<<(NNZ * HEADS + 255) / 256, 256, 0, stream>>>(he_node, he_edge, alpha, amax,
                                                             asum, Dcnt, Bcnt);
  scatter_edges<<<NNZ, dim3(HID, HEADS), 0, stream>>>(he_node, he_edge, alpha, asum, Bcnt, xt, out_e);
  scatter_nodes<<<NNZ, dim3(HID, HEADS), 0, stream>>>(he_node, he_edge, alpha, asum, Dcnt, out_e, out_n);
  finalize_h<<<(N_NODES * HID + 255) / 256, 256, 0, stream>>>(out_n, bias_cv, ts, first_ts, hbuf);

  // -------- BiLSTM layer 1 --------
  gemm_wmma_bf16<<<dim3(GATES / 128, N_NODES / 128), 256, 0, stream>>>(
      hbuf, Wih1f, cbias + 0 * GATES, pre_f, N_NODES, GATES, HID, 0);
  gemm_wmma_bf16<<<dim3(GATES / 128, N_NODES / 128), 256, 0, stream>>>(
      hbuf, Wih1b, cbias + 1 * GATES, pre_b, N_NODES, GATES, HID, 0);
  lstm_layer<<<dim3(BATCH / 16, 2), 512, 0, stream>>>(
      pre_f, pre_b, whhB + 0ull * GATES * RNN_H, whhB + 1ull * GATES * RNN_H, rnn1);

  // -------- BiLSTM layer 2 --------
  gemm_wmma_bf16<<<dim3(GATES / 128, N_NODES / 128), 256, 0, stream>>>(
      rnn1, Wih2f, cbias + 2 * GATES, pre_f, N_NODES, GATES, 2 * RNN_H, 0);
  gemm_wmma_bf16<<<dim3(GATES / 128, N_NODES / 128), 256, 0, stream>>>(
      rnn1, Wih2b, cbias + 3 * GATES, pre_b, N_NODES, GATES, 2 * RNN_H, 0);
  lstm_layer<<<dim3(BATCH / 16, 2), 512, 0, stream>>>(
      pre_f, pre_b, whhB + 2ull * GATES * RNN_H, whhB + 3ull * GATES * RNN_H, rnn2);

  // -------- causal head + roles --------
  gemm_wmma_bf16<<<dim3(HID / 128, N_NODES / 128), 256, 0, stream>>>(
      rnn2, causal_W, causal_b, causal, N_NODES, HID, 2 * RNN_H, 1);
  role_add<<<(N_NODES * HID + 255) / 256, 256, 0, stream>>>(causal, role_emb, roles);

  // -------- output projection (2 GB write, bandwidth bound) --------
  gemm_wmma_bf16<<<dim3(16384 / 128, N_NODES / 128), 256, 0, stream>>>(
      causal, emb_W, emb_b, Cout, N_NODES, 16384, HID, 0);
}
